// HybridBlock_26817775796997
// MI455X (gfx1250) — matrix operand order repaired and checked by reading, not compiled, not run on hardware
//
#include <hip/hip_runtime.h>
#include <hip/hip_bf16.h>
#include <cstdint>

#define DIM     512
#define D_INNER 512
#define DH      256
#define N_STATE 16
#define DT_RANK 32
#define HIDDEN  2048
#define BATCH   4
#define SEQ     4096
#define NTOK    (BATCH*SEQ)   // 16384

typedef unsigned short u16;
typedef __attribute__((ext_vector_type(8)))  float          v8f;
typedef __attribute__((ext_vector_type(16))) __bf16         v16bf;
typedef __attribute__((ext_vector_type(8)))  unsigned short us8;
typedef __attribute__((ext_vector_type(16))) unsigned short us16;

// ---- fp32 -> bf16 round-to-nearest-even --------------------------------
__device__ __forceinline__ u16 f2bf(float f){
  unsigned u = __builtin_bit_cast(unsigned, f);
  u += 0x7FFFu + ((u >> 16) & 1u);
  return (u16)(u >> 16);
}

// ---- CDNA5 async global->LDS staging (ASYNCcnt tracked) ----------------
__device__ __forceinline__ void async_ld_b128(const u16* lds, const u16* g){
  // LDS destination offset comes from a VGPR: low 32 bits of the LDS pointer
  asm volatile("global_load_async_to_lds_b128 %0, %1, off"
               :: "v"((unsigned)(uintptr_t)lds), "v"(g) : "memory");
}
__device__ __forceinline__ void wait_async3(){
  asm volatile("s_wait_asynccnt 0x3" ::: "memory");
}
__device__ __forceinline__ void wait_async0(){
  asm volatile("s_wait_asynccnt 0x0" ::: "memory");
}

// ---- strided fp32 -> bf16 conversion (weights, dt_low slice) -----------
__global__ void cvt2d_f32_bf16(const float* __restrict__ src, u16* __restrict__ dst,
                               int rows, int cols, int src_ld){
  int idx = blockIdx.x * 256 + threadIdx.x;
  if (idx >= rows * cols) return;
  int r = idx / cols, c = idx - r * cols;
  dst[idx] = f2bf(src[(size_t)r * src_ld + c]);
}

// ---- LayerNorm over DIM=512, bf16 output (GEMM operand) ----------------
__global__ void layernorm_bf16(const float* __restrict__ x, const float* __restrict__ w,
                               const float* __restrict__ b, u16* __restrict__ out){
  __shared__ float red[256];
  int t = blockIdx.x, tid = threadIdx.x;
  const float* row = x + (size_t)t * DIM;
  float v0 = row[tid], v1 = row[tid + 256];
  red[tid] = v0 + v1; __syncthreads();
  for (int o = 128; o > 0; o >>= 1){ if (tid < o) red[tid] += red[tid + o]; __syncthreads(); }
  float mean = red[0] * (1.f / DIM);
  __syncthreads();
  float d0 = v0 - mean, d1 = v1 - mean;
  red[tid] = d0 * d0 + d1 * d1; __syncthreads();
  for (int o = 128; o > 0; o >>= 1){ if (tid < o) red[tid] += red[tid + o]; __syncthreads(); }
  float rstd = rsqrtf(red[0] * (1.f / DIM) + 1e-5f);
  u16* orow = out + (size_t)t * DIM;
  orow[tid]       = f2bf(d0 * rstd * w[tid]       + b[tid]);
  orow[tid + 256] = f2bf(d1 * rstd * w[tid + 256] + b[tid + 256]);
}

// ---- fragment assembly: 16 bf16 from two 8-element LDS reads -----------
__device__ __forceinline__ v16bf frag16(const u16* p0, const u16* p1){
  us16 t;
  us8 lo = *(const us8*)p0;
  us8 hi = *(const us8*)p1;
  #pragma unroll
  for (int i = 0; i < 8; i++){ t[i] = lo[i]; t[i + 8] = hi[i]; }
  return __builtin_bit_cast(v16bf, t);
}

// ---- WMMA GEMM: C[M,N] = A[M,K](bf16) . W[N,K]^T(bf16) -----------------
// Block tile 128x64, 8 waves, wave tile 32x32 (4 wmma / K-step, A reuse).
// Double-buffered LDS tiles filled by async global->LDS loads; the last
// K-tile is peeled so the steady-state loop body is branch-free.
// EPI 0: fp32 store          EPI 1: softplus(acc+bias) fp32
// EPI 2: res + gamma*acc     EPI 3: gelu(acc+bias) -> bf16
// EPI 4: res + gamma*(acc+bias)
template<int EPI>
__global__ __launch_bounds__(256) void gemm_bf16_wmma(
    const u16* __restrict__ A, int lda,
    const u16* __restrict__ W, int K, int N,
    float* __restrict__ outF, u16* __restrict__ outB,
    const float* __restrict__ bias,
    const float* __restrict__ res,
    const float* __restrict__ gamma)
{
  __shared__ u16 As[2][128 * 32];   // 2 x 8 KB
  __shared__ u16 Bs[2][64 * 32];    // 2 x 4 KB
  const int tid  = threadIdx.x;
  const int lane = tid & 31;
  const int wave = tid >> 5;
  const int wm = wave >> 1;           // 4 waves along M (32 rows each)
  const int wn = wave & 1;            // 2 waves along N (32 cols each)
  const int m0 = blockIdx.x * 128;
  const int n0 = blockIdx.y * 64;

  v8f acc00 = {}; v8f acc01 = {};     // [i = M-subtile][s = N-subtile]
  v8f acc10 = {}; v8f acc11 = {};

  // async staging pattern: 8-half (16 B) chunks, 4 chunks per 32-wide row
  const int aRow0 = tid >> 2;               // A chunk 0 row (0..63)
  const int aRow1 = (tid + 256) >> 2;       // A chunk 1 row (64..127)
  const int kkA   = (tid & 3) * 8;
  const int bRow  = tid >> 2;               // B row (0..63)

  const int T = K / 32;

  auto issue = [&](int buf, int k0){
    async_ld_b128(&As[buf][(size_t)(aRow0)*32 + kkA],
                  &A[(size_t)(m0 + aRow0) * lda + k0 + kkA]);
    async_ld_b128(&As[buf][(size_t)(aRow1)*32 + kkA],
                  &A[(size_t)(m0 + aRow1) * lda + k0 + kkA]);
    async_ld_b128(&Bs[buf][(size_t)(bRow)*32 + kkA],
                  &W[(size_t)(n0 + bRow) * K + k0 + kkA]);
  };

  const int mr = lane & 15, sel = lane >> 4;    // A-frag lane mapping
  const int nc = lane & 15, ks  = lane >> 4;    // B-frag lane mapping

  auto compute = [&](int cur){
    // A fragments: rows wm*32 + {0,16} + mr ; lane-half selects K octets
    const u16* ap0 = &As[cur][(wm * 32 +      mr) * 32];
    const u16* ap1 = &As[cur][(wm * 32 + 16 + mr) * 32];
    v16bf av0 = frag16(ap0 + sel * 8, ap0 + 16 + sel * 8);
    v16bf av1 = frag16(ap1 + sel * 8, ap1 + 16 + sel * 8);

    // B fragments: cols wn*32 + {0,16} + nc ; lane-half selects K octets
    const u16* bp0 = &Bs[cur][(wn * 32 +      nc) * 32 + ks * 8];
    const u16* bp1 = &Bs[cur][(wn * 32 + 16 + nc) * 32 + ks * 8];
    v16bf bv0 = frag16(bp0, bp0 + 16);
    v16bf bv1 = frag16(bp1, bp1 + 16);

    acc00 = __builtin_amdgcn_wmma_f32_16x16x32_bf16(false, av0, false, bv0, (short)0, acc00, false, false);
    acc01 = __builtin_amdgcn_wmma_f32_16x16x32_bf16(false, av0, false, bv1, (short)0, acc01, false, false);
    acc10 = __builtin_amdgcn_wmma_f32_16x16x32_bf16(false, av1, false, bv0, (short)0, acc10, false, false);
    acc11 = __builtin_amdgcn_wmma_f32_16x16x32_bf16(false, av1, false, bv1, (short)0, acc11, false, false);
  };

  issue(0, 0);                                  // prologue: tile 0

  // steady state: branch-free (issue next / wait current / barrier / 4 wmma)
  for (int kt = 0; kt < T - 1; ++kt){
    const int cur = kt & 1;
    issue(1 - cur, (kt + 1) * 32);              // overlap next tile with compute
    wait_async3();                              // in-order: current tile landed
    __syncthreads();                            // tile visible workgroup-wide
    compute(cur);
    __syncthreads();                            // reads done before buffer reuse
  }

  // peeled last tile: drain ASYNCcnt, no trailing barrier needed
  wait_async0();
  __syncthreads();
  compute((T - 1) & 1);

  // D layout: VGPR j -> row j (lanes 0-15) / j+8 (lanes 16-31); col = lane&15
  const int rs = lane >> 4;
  #pragma unroll
  for (int i = 0; i < 2; i++){
    #pragma unroll
    for (int s = 0; s < 2; s++){
      v8f acc = i ? (s ? acc11 : acc10) : (s ? acc01 : acc00);
      #pragma unroll
      for (int j = 0; j < 8; j++){
        int m = m0 + wm * 32 + i * 16 + j + rs * 8;
        int n = n0 + wn * 32 + s * 16 + nc;
        size_t idx = (size_t)m * N + n;
        float v = acc[j];
        if (EPI == 0){
          outF[idx] = v;
        } else if (EPI == 1){
          v += bias[n];
          outF[idx] = (v > 20.f) ? v : log1pf(__expf(v));
        } else if (EPI == 2){
          outF[idx] = res[idx] + gamma[n] * v;
        } else if (EPI == 3){
          v += bias[n];
          outB[idx] = f2bf(0.5f * v * (1.f + erff(v * 0.70710678118f)));
        } else {
          v += bias[n];
          outF[idx] = res[idx] + gamma[n] * v;
        }
      }
    }
  }
}

// ---- depthwise conv width-3 + SiLU; x half -> xt(f32)+xt_bf, z half -> ycat z slot
__global__ void dwconv_silu(const float* __restrict__ xz,
                            const float* __restrict__ cxw, const float* __restrict__ cxb,
                            const float* __restrict__ czw, const float* __restrict__ czb,
                            float* __restrict__ xt, u16* __restrict__ xt_bf,
                            u16* __restrict__ tok_bf)
{
  int idx = blockIdx.x * 256 + threadIdx.x;   // over NTOK*512
  int c = idx & 511;
  int t = idx >> 9;
  int l = t & (SEQ - 1);
  int ch = (c < DH) ? c : (c - DH);
  const float* wv; float bv;
  if (c < DH){ wv = cxw + ch * 3; bv = cxb[ch]; } else { wv = czw + ch * 3; bv = czb[ch]; }
  float xm = (l > 0)       ? xz[(size_t)(t - 1) * 512 + c] : 0.f;
  float xc =                 xz[(size_t)t       * 512 + c];
  float xp = (l < SEQ - 1) ? xz[(size_t)(t + 1) * 512 + c] : 0.f;
  float v = xm * wv[0] + xc * wv[1] + xp * wv[2] + bv;
  float sv = v / (1.f + __expf(-v));            // SiLU
  if (c < DH){
    xt[(size_t)t * DH + c]    = sv;
    xt_bf[(size_t)t * DH + c] = f2bf(sv);
  } else {
    tok_bf[(size_t)t * 512 + c] = f2bf(sv);     // ycat[:, 256:512]
  }
}

// ---- selective scan: lane = (batch, channel); 16 fp32 states in VGPRs ---
__global__ void selective_scan(const float* __restrict__ xt,
                               const float* __restrict__ delta,
                               const float* __restrict__ xdbl,
                               const float* __restrict__ A_log,
                               const float* __restrict__ Dv,
                               u16* __restrict__ tok_bf)
{
  int b = blockIdx.x;        // 0..BATCH-1
  int d = threadIdx.x;       // 0..DH-1
  float Ac[N_STATE];
  #pragma unroll
  for (int n = 0; n < N_STATE; n++) Ac[n] = -__expf(A_log[d * N_STATE + n]);
  float Dd = Dv[d];
  float h[N_STATE];
  #pragma unroll
  for (int n = 0; n < N_STATE; n++) h[n] = 0.f;

  for (int l = 0; l < SEQ; l++){
    size_t t = (size_t)b * SEQ + l;
    float dlt = delta[t * DH + d];
    float u   = xt[t * DH + d];
    const float* bc = &xdbl[t * 64 + DT_RANK];  // B[16] then C[16]
    float du = dlt * u;
    float y  = u * Dd;
    #pragma unroll
    for (int n = 0; n < N_STATE; n++){
      h[n] = __expf(dlt * Ac[n]) * h[n] + du * bc[n];
      y += h[n] * bc[N_STATE + n];
    }
    tok_bf[t * 512 + d] = f2bf(y);              // ycat[:, 0:256]
  }
}

// ------------------------------------------------------------------------
extern "C" void kernel_launch(void* const* d_in, const int* in_sizes, int n_in,
                              void* d_out, int out_size, void* d_ws, size_t ws_size,
                              hipStream_t stream)
{
  const float* x      = (const float*)d_in[0];
  const float* ln1_w  = (const float*)d_in[1];
  const float* ln1_b  = (const float*)d_in[2];
  const float* w_in   = (const float*)d_in[3];
  const float* cxw    = (const float*)d_in[4];
  const float* cxb    = (const float*)d_in[5];
  const float* czw    = (const float*)d_in[6];
  const float* czb    = (const float*)d_in[7];
  const float* xprw   = (const float*)d_in[8];
  const float* dtpw   = (const float*)d_in[9];
  const float* dtpb   = (const float*)d_in[10];
  const float* A_log  = (const float*)d_in[11];
  const float* Dvec   = (const float*)d_in[12];
  const float* w_out  = (const float*)d_in[13];
  const float* ln2_w  = (const float*)d_in[14];
  const float* ln2_b  = (const float*)d_in[15];
  const float* mlp_w1 = (const float*)d_in[16];
  const float* mlp_b1 = (const float*)d_in[17];
  const float* mlp_w2 = (const float*)d_in[18];
  const float* mlp_b2 = (const float*)d_in[19];
  const float* gamma1 = (const float*)d_in[20];
  const float* gamma2 = (const float*)d_in[21];
  float* out = (float*)d_out;

  char* ws = (char*)d_ws;
  size_t off = 0;
  auto alloc = [&](size_t bytes)->char*{
    char* p = ws + off; off += (bytes + 255) & ~(size_t)255; return p;
  };
  u16*   tok_bf  = (u16*)  alloc((size_t)NTOK * 512 * 2);   // h -> ycat -> h2 (reused)
  float* xz      = (float*)alloc((size_t)NTOK * 512 * 4);
  float* xt      = (float*)alloc((size_t)NTOK * DH  * 4);
  u16*   xt_bf   = (u16*)  alloc((size_t)NTOK * DH  * 2);
  float* xdbl    = (float*)alloc((size_t)NTOK * 64  * 4);
  u16*   dtlo_bf = (u16*)  alloc((size_t)NTOK * DT_RANK * 2);
  float* delta   = (float*)alloc((size_t)NTOK * DH  * 4);
  float* x1      = (float*)alloc((size_t)NTOK * DIM * 4);
  u16*   hid_bf  = (u16*)  alloc((size_t)NTOK * HIDDEN * 2);
  u16*   win_bf  = (u16*)  alloc((size_t)D_INNER * DIM * 2);
  u16*   xprw_bf = (u16*)  alloc((size_t)64 * DH * 2);
  u16*   dtpw_bf = (u16*)  alloc((size_t)DH * DT_RANK * 2);
  u16*   wout_bf = (u16*)  alloc((size_t)DIM * D_INNER * 2);
  u16*   w1_bf   = (u16*)  alloc((size_t)HIDDEN * DIM * 2);
  u16*   w2_bf   = (u16*)  alloc((size_t)DIM * HIDDEN * 2);

  auto cvt = [&](const float* s, u16* dst, int rows, int cols, int ld){
    int total = rows * cols;
    cvt2d_f32_bf16<<<(total + 255) / 256, 256, 0, stream>>>(s, dst, rows, cols, ld);
  };
  cvt(w_in,   win_bf,  D_INNER, DIM,     DIM);
  cvt(xprw,   xprw_bf, 64,      DH,      DH);
  cvt(dtpw,   dtpw_bf, DH,      DT_RANK, DT_RANK);
  cvt(w_out,  wout_bf, DIM,     D_INNER, D_INNER);
  cvt(mlp_w1, w1_bf,   HIDDEN,  DIM,     DIM);
  cvt(mlp_w2, w2_bf,   DIM,     HIDDEN,  HIDDEN);

  // LN1 -> h (bf16)
  layernorm_bf16<<<NTOK, 256, 0, stream>>>(x, ln1_w, ln1_b, tok_bf);

  // xz = h @ w_in^T
  { dim3 g(NTOK / 128, D_INNER / 64);
    gemm_bf16_wmma<0><<<g, 256, 0, stream>>>(tok_bf, DIM, win_bf, DIM, D_INNER,
                                             xz, nullptr, nullptr, nullptr, nullptr); }

  // depthwise conv + SiLU (x half -> xt; z half -> ycat slot)
  dwconv_silu<<<(NTOK * 512) / 256, 256, 0, stream>>>(xz, cxw, cxb, czw, czb,
                                                      xt, xt_bf, tok_bf);

  // x_dbl = xt @ x_proj^T   (N = 64: dt_low | B | C)
  { dim3 g(NTOK / 128, 64 / 64);
    gemm_bf16_wmma<0><<<g, 256, 0, stream>>>(xt_bf, DH, xprw_bf, DH, 64,
                                             xdbl, nullptr, nullptr, nullptr, nullptr); }

  // dt_low slice -> bf16
  cvt(xdbl, dtlo_bf, NTOK, DT_RANK, 64);

  // delta = softplus(dt_low @ dt_proj^T + dt_proj_b)
  { dim3 g(NTOK / 128, DH / 64);
    gemm_bf16_wmma<1><<<g, 256, 0, stream>>>(dtlo_bf, DT_RANK, dtpw_bf, DT_RANK, DH,
                                             delta, nullptr, dtpb, nullptr, nullptr); }

  // selective scan -> ycat y-half (bf16)
  selective_scan<<<BATCH, DH, 0, stream>>>(xt, delta, xdbl, A_log, Dvec, tok_bf);

  // x1 = x + gamma1 * (ycat @ w_out^T)
  { dim3 g(NTOK / 128, DIM / 64);
    gemm_bf16_wmma<2><<<g, 256, 0, stream>>>(tok_bf, D_INNER, wout_bf, D_INNER, DIM,
                                             x1, nullptr, nullptr, x, gamma1); }

  // LN2 -> h2 (bf16)
  layernorm_bf16<<<NTOK, 256, 0, stream>>>(x1, ln2_w, ln2_b, tok_bf);

  // hid = gelu(h2 @ mlp_w1^T + b1)  (bf16)
  { dim3 g(NTOK / 128, HIDDEN / 64);
    gemm_bf16_wmma<3><<<g, 256, 0, stream>>>(tok_bf, DIM, w1_bf, DIM, HIDDEN,
                                             nullptr, hid_bf, mlp_b1, nullptr, nullptr); }

  // out = x1 + gamma2 * (hid @ mlp_w2^T + b2)
  { dim3 g(NTOK / 128, DIM / 64);
    gemm_bf16_wmma<4><<<g, 256, 0, stream>>>(hid_bf, HIDDEN, w2_bf, HIDDEN, DIM,
                                             out, nullptr, mlp_b2, x1, gamma2); }

  (void)in_sizes; (void)n_in; (void)out_size; (void)ws_size;
}
